// Model_39135742001730
// MI455X (gfx1250) — matrix operand order repaired and checked by reading, not compiled, not run on hardware
//
#include <hip/hip_runtime.h>

#define B_  256
#define T_  720
#define I_  16
#define H_  128
#define PL_ 96

typedef __attribute__((ext_vector_type(16))) _Float16 v16h;
typedef __attribute__((ext_vector_type(8)))  _Float16 v8h;
typedef __attribute__((ext_vector_type(8)))  float    v8f;

// ---- WMMA operand builders (layouts per CDNA5 ISA 7.12.2) -------------------

// B operand (K=32 x N=16) from fp32 weight matrix W (row n, ld floats, col k):
// B[k][n] = W[n][k]; lane ln supplies column n0+ln; half h: K=8h+[0..7], 16+8h+[0..7].
__device__ __forceinline__ v16h b_tile_f32(const float* W, int ld, int n0, int k0,
                                           int ln, int hlf) {
  const float* p = W + (size_t)(n0 + ln) * ld + k0 + hlf * 8;
  v16h r;
#pragma unroll
  for (int i = 0; i < 16; ++i) r[i] = (_Float16)p[i < 8 ? i : i + 8];
  return r;
}

// B operand for W_ih with bias folded in (true K=16, padded to 32, where the
// K=16 row carries the bias; the matching A operand has 1.0 at K=16).
// half h: elems 0..7 = W[n][8h+i]; elem 8 = bias[n] (K=16) for h0, 0 for h1; rest 0.
__device__ __forceinline__ v16h b_tile_ih(const float* W, const float* bv,
                                          int n0, int ln, int hlf) {
  v16h r;
  const int n = n0 + ln;
  const float* p = W + (size_t)n * I_ + hlf * 8;
#pragma unroll
  for (int i = 0; i < 8; ++i) r[i] = (_Float16)p[i];
  r[8] = (hlf == 0) ? (_Float16)bv[n] : (_Float16)0.0f;
#pragma unroll
  for (int i = 9; i < 16; ++i) r[i] = (_Float16)0.0f;
  return r;
}

// A operand (M=16 x K=32) from f16 row-major LDS buffer (row M=ln, stride ld
// halfs). half h: first 8 halfs K=k0+8h+[0..7], last 8 halfs K=k0+16+8h+[0..7].
__device__ __forceinline__ v16h a_tile_h(const _Float16* hb, int ld, int k0,
                                         int ln, int hlf) {
  const _Float16* p = hb + ln * ld + k0 + hlf * 8;
  v8h lo = *(const v8h*)p;
  v8h hi = *(const v8h*)(p + 16);
  v16h r;
#pragma unroll
  for (int i = 0; i < 8; ++i) { r[i] = lo[i]; r[i + 8] = hi[i]; }
  return r;
}

// A operand for x (K=16 padded to 32, with K=16 element = 1.0 for the bias row)
// from fp32 global row. half0: K=0..7 data, K=16 -> 1.0; half1: K=8..15 data.
__device__ __forceinline__ v16h a_tile_x_g(const float* xrow, int hlf) {
  const float* p = xrow + hlf * 8;
  v16h r;
#pragma unroll
  for (int i = 0; i < 8; ++i) r[i] = (_Float16)p[i];
  r[8] = (hlf == 0) ? (_Float16)1.0f : (_Float16)0.0f;
#pragma unroll
  for (int i = 9; i < 16; ++i) r[i] = (_Float16)0.0f;
  return r;
}

// Same, from f16 LDS buffer with stride 24 halfs.
__device__ __forceinline__ v16h a_tile_x_l(const _Float16* xb, int ln, int hlf) {
  v8h lo = *(const v8h*)(xb + ln * 24 + hlf * 8);
  v16h r;
#pragma unroll
  for (int i = 0; i < 8; ++i) r[i] = lo[i];
  r[8] = (hlf == 0) ? (_Float16)1.0f : (_Float16)0.0f;
#pragma unroll
  for (int i = 9; i < 16; ++i) r[i] = (_Float16)0.0f;
  return r;
}

__device__ __forceinline__ float sigm(float x) { return 1.0f / (1.0f + __expf(-x)); }
__device__ __forceinline__ float tanh_f(float x) {
  float t = __expf(-2.0f * fabsf(x));
  float r = (1.0f - t) / (1.0f + t);
  return copysignf(r, x);
}

// ---------------------------------------------------------------------------
// Encoder: grid = 32 blocks (dir = blk>>4, batch tile = blk&15), 256 threads.
// Wave w owns hidden slice j in [16w,16w+16); its 4 accumulators are the four
// gates (i,f,g,o) for those j -> activations are register-local per lane.
// h is double-buffered in LDS -> exactly ONE barrier per time step.
// ---------------------------------------------------------------------------
__global__ __launch_bounds__(256) void bilstm_encoder(
    const float* __restrict__ x,
    const float* __restrict__ Wih_f, const float* __restrict__ Whh_f,
    const float* __restrict__ b_f,
    const float* __restrict__ Wih_b, const float* __restrict__ Whh_b,
    const float* __restrict__ b_b,
    float* __restrict__ ws)
{
  __shared__ __align__(16) _Float16 hbuf[2][16 * 136];  // 16 x 128 (+8 pad) halfs
  const int tid  = threadIdx.x;
  const int wave = tid >> 5, lane = tid & 31, hlf = lane >> 4, ln = lane & 15;
  const int dir = blockIdx.x >> 4, btile = blockIdx.x & 15;
  const int bbase = btile * 16;
  const float* Wih = dir ? Wih_b : Wih_f;
  const float* Whh = dir ? Whh_b : Whh_f;
  const float* bv  = dir ? b_b  : b_f;

  // Pre-build all B operands in registers for the whole T loop.
  v16h Bhh[4][4], Bih[4];
#pragma unroll
  for (int g = 0; g < 4; ++g) {
    const int n0 = g * H_ + 16 * wave;
#pragma unroll
    for (int kt = 0; kt < 4; ++kt)
      Bhh[g][kt] = b_tile_f32(Whh, H_, n0, kt * 32, ln, hlf);
    Bih[g] = b_tile_ih(Wih, bv, n0, ln, hlf);
  }

  for (int i = tid; i < 16 * 136; i += 256) hbuf[0][i] = (_Float16)0.0f;
  float c[8];
#pragma unroll
  for (int r = 0; r < 8; ++r) c[r] = 0.0f;
  const float* xrow = x + (size_t)(bbase + ln) * T_ * I_;   // A row M = ln
  const int jcol = 16 * wave + ln;
  __syncthreads();

  for (int t = 0; t < T_; ++t) {
    const _Float16* hR = hbuf[t & 1];
    _Float16*       hW = hbuf[(t & 1) ^ 1];
    const int te = dir ? (T_ - 1 - t) : t;
    const v8f zacc = {};
    v8f acc[4];

    { // x @ W_ih^T + bias  (K=16 data + bias row, padded to 32; C = 0)
      v16h ax = a_tile_x_g(xrow + (size_t)te * I_, hlf);
#pragma unroll
      for (int g = 0; g < 4; ++g)
        acc[g] = __builtin_amdgcn_wmma_f32_16x16x32_f16(
            false, ax, false, Bih[g], (short)0, zacc, false, false);
    }
    // h @ W_hh^T  (K=128 as 4 K-tiles)
#pragma unroll
    for (int kt = 0; kt < 4; ++kt) {
      v16h ah = a_tile_h(hR, 136, kt * 32, ln, hlf);
#pragma unroll
      for (int g = 0; g < 4; ++g)
        acc[g] = __builtin_amdgcn_wmma_f32_16x16x32_f16(
            false, ah, false, Bhh[g][kt], (short)0, acc[g], false, false);
    }

    // LSTM cell update; C/D layout: M = r + 8*hlf, N(=j) = jcol
#pragma unroll
    for (int r = 0; r < 8; ++r) {
      float iv = sigm(acc[0][r]);
      float fv = sigm(acc[1][r]);
      float gv = tanh_f(acc[2][r]);
      float ov = sigm(acc[3][r]);
      c[r] = fv * c[r] + iv * gv;
      float hv = ov * tanh_f(c[r]);
      hW[(r + 8 * hlf) * 136 + jcol] = (_Float16)hv;
    }
    __syncthreads();   // new h visible; old buffer free to overwrite next step
  }

  // Final (h, c) -> workspace: [hf | cf | hb | cb], each B_*H_ floats.
  // T_ is even -> final h lives in hbuf[0]; this lane reads back its own writes.
  float* hOut = ws + (size_t)dir * 2 * B_ * H_;
  float* cOut = hOut + (size_t)B_ * H_;
#pragma unroll
  for (int r = 0; r < 8; ++r) {
    const int b = bbase + r + 8 * hlf;
    hOut[(size_t)b * H_ + jcol] = (float)hbuf[0][(r + 8 * hlf) * 136 + jcol];
    cOut[(size_t)b * H_ + jcol] = c[r];
  }
}

// ---------------------------------------------------------------------------
// Decoder: 16 blocks x 512 threads (16 waves: waves 0-7 = forward cell,
// 8-15 = backward cell). cat = [hf | hb] double-buffered in LDS feeds both
// cells and the 16x16x256 output projection (wave 0). Two barriers per step.
// ---------------------------------------------------------------------------
__global__ __launch_bounds__(512) void bilstm_decoder(
    const float* __restrict__ x,
    const float* __restrict__ Wih_f, const float* __restrict__ Whh_f,
    const float* __restrict__ b_f,
    const float* __restrict__ Wih_b, const float* __restrict__ Whh_b,
    const float* __restrict__ b_b,
    const float* __restrict__ Wlin, const float* __restrict__ blin,
    const float* __restrict__ ws, float* __restrict__ out)
{
  __shared__ __align__(16) _Float16 cat[2][16 * 264];  // 16 x 256 (+8 pad) halfs
  __shared__ __align__(16) _Float16 xb2[2][16 * 24];   // 16 x 16 (+8 pad) halfs
  const int tid  = threadIdx.x;
  const int wave = tid >> 5, lane = tid & 31, hlf = lane >> 4, ln = lane & 15;
  const int cell = wave >> 3, w8 = wave & 7;
  const int bbase = blockIdx.x * 16;
  const float* Wih = cell ? Wih_b : Wih_f;
  const float* Whh = cell ? Whh_b : Whh_f;
  const float* bv  = cell ? b_b  : b_f;

  v16h Bhh[4][4], Bih[4];
#pragma unroll
  for (int g = 0; g < 4; ++g) {
    const int n0 = g * H_ + 16 * w8;
#pragma unroll
    for (int kt = 0; kt < 4; ++kt)
      Bhh[g][kt] = b_tile_f32(Whh, H_, n0, kt * 32, ln, hlf);
    Bih[g] = b_tile_ih(Wih, bv, n0, ln, hlf);
  }
  const float blv = blin[ln];

  // c state from encoder workspace (cf at +BH, cb at +3BH)
  const float* cIn = ws + (size_t)B_ * H_ + (size_t)cell * 2 * B_ * H_;
  const int jcol = 16 * w8 + ln;
  float c[8];
#pragma unroll
  for (int r = 0; r < 8; ++r)
    c[r] = cIn[(size_t)(bbase + r + 8 * hlf) * H_ + jcol];

  // cat[0] <- [hf | hb] as f16
  for (int idx = tid; idx < 16 * 256; idx += 512) {
    int row = idx >> 8, col = idx & 255;
    const float* hIn = ws + (col < 128 ? (size_t)0 : (size_t)2 * B_ * H_);
    cat[0][row * 264 + col] = (_Float16)hIn[(size_t)(bbase + row) * H_ + (col & 127)];
  }
  // first decoder input = x[:, T-1, :]
  for (int idx = tid; idx < 16 * 16; idx += 512) {
    int row = idx >> 4, col = idx & 15;
    xb2[0][row * 24 + col] =
        (_Float16)x[((size_t)(bbase + row) * T_ + (T_ - 1)) * I_ + col];
  }
  __syncthreads();

  for (int s = 0; s < PL_; ++s) {
    const int rb = s & 1, wb = rb ^ 1;
    const v8f zacc = {};
    v8f acc[4];

    { // input contribution (+bias via augmented K row, C = 0)
      v16h ax = a_tile_x_l(xb2[rb], ln, hlf);
#pragma unroll
      for (int g = 0; g < 4; ++g)
        acc[g] = __builtin_amdgcn_wmma_f32_16x16x32_f16(
            false, ax, false, Bih[g], (short)0, zacc, false, false);
    }
    // recurrent contribution (this cell's half of cat)
#pragma unroll
    for (int kt = 0; kt < 4; ++kt) {
      v16h ah = a_tile_h(cat[rb] + cell * 128, 264, kt * 32, ln, hlf);
#pragma unroll
      for (int g = 0; g < 4; ++g)
        acc[g] = __builtin_amdgcn_wmma_f32_16x16x32_f16(
            false, ah, false, Bhh[g][kt], (short)0, acc[g], false, false);
    }
#pragma unroll
    for (int r = 0; r < 8; ++r) {
      float iv = sigm(acc[0][r]);
      float fv = sigm(acc[1][r]);
      float gv = tanh_f(acc[2][r]);
      float ov = sigm(acc[3][r]);
      c[r] = fv * c[r] + iv * gv;
      float hv = ov * tanh_f(c[r]);
      cat[wb][(r + 8 * hlf) * 264 + cell * 128 + jcol] = (_Float16)hv;
    }
    __syncthreads();   // full new [hf|hb] (cat[wb]) visible

    if (wave == 0) {   // wave-uniform branch: EXEC all ones inside
      v8f oacc;
#pragma unroll
      for (int i = 0; i < 8; ++i) oacc[i] = blv;
#pragma unroll
      for (int kt = 0; kt < 8; ++kt) {
        v16h a  = a_tile_h(cat[wb], 264, kt * 32, ln, hlf);
        v16h bl = b_tile_f32(Wlin, 2 * H_, 0, kt * 32, ln, hlf);
        oacc = __builtin_amdgcn_wmma_f32_16x16x32_f16(
            false, a, false, bl, (short)0, oacc, false, false);
      }
#pragma unroll
      for (int r = 0; r < 8; ++r) {
        const int row = r + 8 * hlf;
        float v = oacc[r];
        out[((size_t)(bbase + row) * PL_ + s) * I_ + ln] = v;   // (B, PL, I)
        xb2[wb][row * 24 + ln] = (_Float16)v;                   // next input
      }
    }
    __syncthreads();
  }
}

extern "C" void kernel_launch(void* const* d_in, const int* in_sizes, int n_in,
                              void* d_out, int out_size, void* d_ws, size_t ws_size,
                              hipStream_t stream) {
  const float* x      = (const float*)d_in[0];
  const float* eWih_f = (const float*)d_in[1];
  const float* eWhh_f = (const float*)d_in[2];
  const float* eb_f   = (const float*)d_in[3];
  const float* eWih_b = (const float*)d_in[4];
  const float* eWhh_b = (const float*)d_in[5];
  const float* eb_b   = (const float*)d_in[6];
  const float* dWih_f = (const float*)d_in[7];
  const float* dWhh_f = (const float*)d_in[8];
  const float* db_f   = (const float*)d_in[9];
  const float* dWih_b = (const float*)d_in[10];
  const float* dWhh_b = (const float*)d_in[11];
  const float* db_b   = (const float*)d_in[12];
  const float* Wlin   = (const float*)d_in[13];
  const float* blin   = (const float*)d_in[14];
  float* ws = (float*)d_ws;

  bilstm_encoder<<<32, 256, 0, stream>>>(x, eWih_f, eWhh_f, eb_f,
                                         eWih_b, eWhh_b, eb_b, ws);
  bilstm_decoder<<<16, 512, 0, stream>>>(x, dWih_f, dWhh_f, db_f,
                                         dWih_b, dWhh_b, db_b,
                                         Wlin, blin, ws, (float*)d_out);
}
